// AttentionPool_35287451304890
// MI455X (gfx1250) — matrix operand order repaired and checked by reading, not compiled, not run on hardware
//
#include <hip/hip_runtime.h>
#include <hip/hip_bf16.h>
#include <math.h>

// ---------------------------------------------------------------------------
// AttentionPool on MI455X (gfx1250, wave32, WMMA)
//
// Algebraic restructuring (R=8 queries only) turns ~310 GFLOP of projections
// into ~40 GFLOP of WMMA work that is HBM-bound on reading x twice (~400MB
// -> ~17us at 23.3 TB/s):
//   qk[b,h*8+r,c] = sum_d q[b,r,h,d]*Wk[h*64+d,c]         (tiny fold)
//   logits[b,m,n] = qk[b,m,:] . x[b,n,:]                  (WMMA GEMM #1)
//   attn          = softmax(mask(logits))                 (wave-per-row)
//   y[b,m,c]      = sum_n attn[b,m,n] * x[b,n,c]          (WMMA GEMM #2)
//   x_cls[b,r,h*64+d] = Wv[h*64+d,:] . y[b,h*8+r,:]       (tiny)
//   out = x_cls @ Wp.T + bp                               (tiny)
//
// Small kernels are cache-blocked (16x16 tiles / batch-looped) so weight
// matrices are read ~once from L2 instead of once per activation row.
// bf16 WMMA inputs produced in-register (v_cvt_pk_bf16_f32), fp32 accumulate.
// ---------------------------------------------------------------------------

#define DIM    1536
#define NTOK   4096
#define HEADS  24
#define RPT    8
#define MROWS  192              // HEADS * RPT
#define NBATCH 8
#define SCALE  0.125f           // 64^-0.5

typedef __bf16 bf16_t;
typedef __attribute__((ext_vector_type(16))) __bf16 v16bf;
typedef __attribute__((ext_vector_type(8)))  __bf16 v8bf;
typedef __attribute__((ext_vector_type(8)))  float  v8f;
typedef __attribute__((ext_vector_type(4)))  float  v4f;

static __device__ __forceinline__ v16bf cat8(v8bf lo, v8bf hi) {
  return __builtin_shufflevector(lo, hi, 0,1,2,3,4,5,6,7,8,9,10,11,12,13,14,15);
}

// ------------------------- small dense kernels -----------------------------

// q[b*8+r][cp] = SCALE * x[b, r, :] . Wq[cp, :]
// block = 16 cp x 16 t: weight rows shared by 16 t-threads, x broadcast.
__global__ __launch_bounds__(256)
void qproj_kernel(const float* __restrict__ x, const float* __restrict__ Wq,
                  float* __restrict__ q) {
  int cp = blockIdx.x * 16 + (threadIdx.x & 15);   // 0..1535
  int t  = blockIdx.y * 16 + (threadIdx.x >> 4);   // 0..63
  int b  = t >> 3, r = t & 7;
  const float* xr = x  + ((size_t)b * NTOK + r) * DIM;
  const float* wr = Wq + (size_t)cp * DIM;
  float acc = 0.f;
  for (int c = 0; c < DIM; c += 4) {
    v4f xa = *(const v4f*)(xr + c);
    v4f wa = *(const v4f*)(wr + c);
    acc += xa[0]*wa[0] + xa[1]*wa[1] + xa[2]*wa[2] + xa[3]*wa[3];
  }
  q[(size_t)t * DIM + cp] = acc * SCALE;
}

// qkb[(b*192 + h*8 + r)*DIM + c] = bf16( sum_d q[b*8+r][h*64+d] * Wk[h*64+d][c] )
// One thread computes all 8 batches for its (h, r, c): Wk read exactly once.
__global__ __launch_bounds__(256)
void fold_kernel(const float* __restrict__ q, const float* __restrict__ Wk,
                 bf16_t* __restrict__ qkb) {
  int c = blockIdx.x * 32 + (threadIdx.x & 31);    // 0..1535
  int r = threadIdx.x >> 5;                        // 0..7
  int h = blockIdx.y;                              // 0..23
  float acc[NBATCH] = {};
  #pragma unroll 4
  for (int d = 0; d < 64; ++d) {
    float wk = Wk[(size_t)(h * 64 + d) * DIM + c];
    #pragma unroll
    for (int b = 0; b < NBATCH; ++b)
      acc[b] += q[((size_t)b * RPT + r) * DIM + h * 64 + d] * wk;
  }
  #pragma unroll
  for (int b = 0; b < NBATCH; ++b)
    qkb[((size_t)b * MROWS + h * RPT + r) * DIM + c] = (bf16_t)acc[b];
}

// ------------------------- GEMM #1: logits ---------------------------------
// logits[b][m][n] = qk[b][m][:] . x[b][n][:]   (M=192, N=4096, K=1536)
__global__ __launch_bounds__(256)
void gemm1_logits(const bf16_t* __restrict__ qkb, const float* __restrict__ x,
                  float* __restrict__ logits) {
  const int b    = blockIdx.y;
  const int n0   = blockIdx.x * 64;
  const int tid  = threadIdx.x;
  const int lane = tid & 31;
  const int w    = tid >> 5;
  const int mg   = w & 3;        // 4 groups x 3 M-tiles = 12 tiles (M=192)
  const int ng   = w >> 2;       // 2 groups x 2 N-tiles = 4 tiles (N=64)
  const int lh   = lane >> 4;    // lane half
  const int l16  = lane & 15;

  const bf16_t* qk_b  = qkb    + (size_t)b * MROWS * DIM;
  const float*  x_b   = x      + (size_t)b * NTOK  * DIM;
  float*        out_b = logits + (size_t)b * MROWS * NTOK;

  v8f acc[3][2] = {};

  for (int k0 = 0; k0 < DIM; k0 += 32) {
    v16bf a[3];
    #pragma unroll
    for (int i = 0; i < 3; ++i) {
      const bf16_t* row = qk_b + (size_t)((mg*3+i)*16 + l16) * DIM + k0 + lh*16;
      a[i] = cat8(*(const v8bf*)row, *(const v8bf*)(row + 8));
    }
    v16bf bf[2];
    #pragma unroll
    for (int j = 0; j < 2; ++j) {
      const float* xp = x_b + (size_t)(n0 + (ng*2+j)*16 + l16) * DIM + k0 + lh*16;
      v4f f0 = *(const v4f*)(xp);
      v4f f1 = *(const v4f*)(xp + 4);
      v4f f2 = *(const v4f*)(xp + 8);
      v4f f3 = *(const v4f*)(xp + 12);
      v16bf t;
      #pragma unroll
      for (int e = 0; e < 4; ++e) {
        t[e]    = (bf16_t)f0[e];  t[4+e]  = (bf16_t)f1[e];
        t[8+e]  = (bf16_t)f2[e];  t[12+e] = (bf16_t)f3[e];
      }
      bf[j] = t;
    }
    #pragma unroll
    for (int i = 0; i < 3; ++i)
      #pragma unroll
      for (int j = 0; j < 2; ++j)
        acc[i][j] = __builtin_amdgcn_wmma_f32_16x16x32_bf16(
            false, a[i], false, bf[j], (short)0, acc[i][j], false, false);
  }

  #pragma unroll
  for (int i = 0; i < 3; ++i)
    #pragma unroll
    for (int j = 0; j < 2; ++j) {
      int n = n0 + (ng*2+j)*16 + l16;
      #pragma unroll
      for (int r = 0; r < 8; ++r) {
        int m = (mg*3+i)*16 + r + lh*8;
        out_b[(size_t)m * NTOK + n] = acc[i][j][r];
      }
    }
}

// ------------------------- mask + softmax ----------------------------------
// one wave per row of logits[b][m][:]; mask: eye(R) ++ mask[b,r,:]
__global__ __launch_bounds__(128)
void mask_softmax(const float* __restrict__ logits,
                  const unsigned char* __restrict__ mask,   // jnp bool = 1B
                  bf16_t* __restrict__ attnb) {
  int row  = blockIdx.x * 4 + (threadIdx.x >> 5);   // 0..1535
  int lane = threadIdx.x & 31;
  int b = row / MROWS, m = row % MROWS, r = m & 7;
  const float* lg = logits + (size_t)row * NTOK;
  bf16_t*      ab = attnb  + (size_t)row * NTOK;
  const unsigned char* mk = mask + ((size_t)b * RPT + r) * (NTOK - RPT);

  float mx = -INFINITY;
  for (int n = lane; n < NTOK; n += 32) {
    bool valid = (n < RPT) ? (n == r) : (mk[n - RPT] != 0);
    if (valid) mx = fmaxf(mx, lg[n]);
  }
  #pragma unroll
  for (int off = 16; off > 0; off >>= 1) mx = fmaxf(mx, __shfl_xor(mx, off, 32));

  float s = 0.f;
  for (int n = lane; n < NTOK; n += 32) {
    bool valid = (n < RPT) ? (n == r) : (mk[n - RPT] != 0);
    if (valid) s += __expf(lg[n] - mx);
  }
  #pragma unroll
  for (int off = 16; off > 0; off >>= 1) s += __shfl_xor(s, off, 32);
  float inv = 1.f / s;

  for (int n = lane; n < NTOK; n += 32) {
    bool valid = (n < RPT) ? (n == r) : (mk[n - RPT] != 0);
    ab[n] = (bf16_t)(valid ? __expf(lg[n] - mx) * inv : 0.f);
  }
}

// ------------------------- GEMM #2: attn @ x -------------------------------
// y[b][m][c] = sum_n attnb[b][m][n] * x[b][n][c]  (M=192, N(=c)=1536, K=4096)
// B-operand (x) needs K-major per lane -> stage 32x64 tile transposed in LDS.
__global__ __launch_bounds__(256)
void gemm2_wsum(const bf16_t* __restrict__ attnb, const float* __restrict__ x,
                float* __restrict__ y) {
  __shared__ bf16_t xt[64][40];   // [c][n], stride 40 -> rows 80B (16B aligned)

  const int b    = blockIdx.y;
  const int c0   = blockIdx.x * 64;
  const int tid  = threadIdx.x;
  const int lane = tid & 31;
  const int w    = tid >> 5;
  const int mg   = w & 3;
  const int cg   = w >> 2;
  const int lh   = lane >> 4;
  const int l16  = lane & 15;

  const bf16_t* at_b = attnb + (size_t)b * MROWS * NTOK;
  const float*  x_b  = x     + (size_t)b * NTOK  * DIM;
  float*        y_b  = y     + (size_t)b * MROWS * DIM;

  const int sn = tid >> 3;          // staging: token row 0..31
  const int sc = (tid & 7) * 8;     // staging: col offset 0..56

  v8f acc[3][2] = {};

  for (int n0 = 0; n0 < NTOK; n0 += 32) {
    const float* xp = x_b + (size_t)(n0 + sn) * DIM + c0 + sc;
    v4f f0 = *(const v4f*)(xp);
    v4f f1 = *(const v4f*)(xp + 4);
    __syncthreads();                               // prev-iter reads done
    #pragma unroll
    for (int e = 0; e < 4; ++e) {
      xt[sc + e][sn]     = (bf16_t)f0[e];
      xt[sc + 4 + e][sn] = (bf16_t)f1[e];
    }
    __syncthreads();

    v16bf a[3];
    #pragma unroll
    for (int i = 0; i < 3; ++i) {
      const bf16_t* row = at_b + (size_t)((mg*3+i)*16 + l16) * NTOK + n0 + lh*16;
      a[i] = cat8(*(const v8bf*)row, *(const v8bf*)(row + 8));
    }
    v16bf bf[2];
    #pragma unroll
    for (int j = 0; j < 2; ++j) {
      const bf16_t* base = &xt[(cg*2+j)*16 + l16][lh * 16];
      bf[j] = cat8(*(const v8bf*)base, *(const v8bf*)(base + 8));
    }
    #pragma unroll
    for (int i = 0; i < 3; ++i)
      #pragma unroll
      for (int j = 0; j < 2; ++j)
        acc[i][j] = __builtin_amdgcn_wmma_f32_16x16x32_bf16(
            false, a[i], false, bf[j], (short)0, acc[i][j], false, false);
  }

  #pragma unroll
  for (int i = 0; i < 3; ++i)
    #pragma unroll
    for (int j = 0; j < 2; ++j) {
      int c = c0 + (cg*2+j)*16 + l16;
      #pragma unroll
      for (int r = 0; r < 8; ++r) {
        int m = (mg*3+i)*16 + r + lh*8;
        y_b[(size_t)m * DIM + c] = acc[i][j][r];
      }
    }
}

// x_cls[t][c] = Wv[c,:] . y[b, (c/64)*8 + r, :]   (t = b*8+r)
// block = 16 c x 16 t (c-chunk stays inside one head since 16 | 64).
__global__ __launch_bounds__(256)
void vproj_kernel(const float* __restrict__ y, const float* __restrict__ Wv,
                  float* __restrict__ xcls) {
  int c = blockIdx.x * 16 + (threadIdx.x & 15);    // 0..1535
  int t = blockIdx.y * 16 + (threadIdx.x >> 4);    // 0..63
  int b = t >> 3, r = t & 7;
  int h = c >> 6;
  const float* yr = y  + ((size_t)b * MROWS + h * RPT + r) * DIM;
  const float* wr = Wv + (size_t)c * DIM;
  float acc = 0.f;
  for (int c2 = 0; c2 < DIM; c2 += 4) {
    v4f ya = *(const v4f*)(yr + c2);
    v4f wa = *(const v4f*)(wr + c2);
    acc += ya[0]*wa[0] + ya[1]*wa[1] + ya[2]*wa[2] + ya[3]*wa[3];
  }
  xcls[(size_t)t * DIM + c] = acc;
}

// out[t][cp] = bp[cp] + x_cls[t,:] . Wp[cp,:]
__global__ __launch_bounds__(256)
void outproj_kernel(const float* __restrict__ xcls, const float* __restrict__ Wp,
                    const float* __restrict__ bp, float* __restrict__ out) {
  int cp = blockIdx.x * 16 + (threadIdx.x & 15);   // 0..1535
  int t  = blockIdx.y * 16 + (threadIdx.x >> 4);   // 0..63
  const float* xr = xcls + (size_t)t * DIM;
  const float* wr = Wp   + (size_t)cp * DIM;
  float acc = 0.f;
  for (int c = 0; c < DIM; c += 4) {
    v4f xa = *(const v4f*)(xr + c);
    v4f wa = *(const v4f*)(wr + c);
    acc += xa[0]*wa[0] + xa[1]*wa[1] + xa[2]*wa[2] + xa[3]*wa[3];
  }
  out[(size_t)t * DIM + cp] = acc + bp[cp];
}

// ---------------------------------------------------------------------------

extern "C" void kernel_launch(void* const* d_in, const int* in_sizes, int n_in,
                              void* d_out, int out_size, void* d_ws, size_t ws_size,
                              hipStream_t stream) {
  const float* x    = (const float*)d_in[0];
  const unsigned char* mask = (const unsigned char*)d_in[1]; // jnp bool -> 1B
  const float* Wq   = (const float*)d_in[2];
  const float* Wk   = (const float*)d_in[3];
  const float* Wv   = (const float*)d_in[4];
  const float* Wp   = (const float*)d_in[5];
  const float* bp   = (const float*)d_in[6];
  float* out        = (float*)d_out;

  // workspace layout (~53 MB total, all 256B aligned)
  char* ws = (char*)d_ws;
  float*  q      = (float*)(ws);                       //  0.39 MB [64][1536]
  bf16_t* qkb    = (bf16_t*)(ws + 393216);             //  4.72 MB [8][192][1536]
  float*  logits = (float*)(ws + 5111808);             // 25.17 MB [8][192][4096]
  bf16_t* attnb  = (bf16_t*)(ws + 30277632);           // 12.58 MB [8][192][4096]
  float*  y      = (float*)(ws + 42860544);            //  9.44 MB [8][192][1536]
  float*  xcls   = (float*)(ws + 52297728);            //  0.39 MB [64][1536]

  qproj_kernel<<<dim3(96, 4), 256, 0, stream>>>(x, Wq, q);
  fold_kernel<<<dim3(48, HEADS), 256, 0, stream>>>(q, Wk, qkb);
  gemm1_logits<<<dim3(64, NBATCH), 256, 0, stream>>>(qkb, x, logits);
  mask_softmax<<<384, 128, 0, stream>>>(logits, mask, attnb);
  gemm2_wsum<<<dim3(24, NBATCH), 256, 0, stream>>>(attnb, x, y);
  vproj_kernel<<<dim3(96, 4), 256, 0, stream>>>(y, Wv, xcls);
  outproj_kernel<<<dim3(96, 4), 256, 0, stream>>>(xcls, Wp, bp, out);
}
